// MambaVisionMixer_30863634989315
// MI455X (gfx1250) — compile-verified, matrix operand order checked by reading, not run on hardware
//
#include <hip/hip_runtime.h>

// ---------------------------------------------------------------------------
// MambaVisionMixer forward for MI455X (gfx1250, wave32, WMMA)
// ---------------------------------------------------------------------------
#define D_MODEL 512
#define D_STATE 16
#define D_CONV  4
#define D_INNER 1024
#define D_HALF  512
#define DT_RANK 32
#define NDBL    64          // DT_RANK + 2*D_STATE
#define B_SZ    4
#define SEQ     4096
#define MROWS   (B_SZ * SEQ)   // 16384
#define LN_EPS  1e-5f
#define SCAN_CHUNK 64

typedef __attribute__((ext_vector_type(16))) __bf16 v16bf;
typedef __attribute__((ext_vector_type(8)))  float  v8f;

// ---- fp32 -> bf16 (round-to-nearest-even), bit-exact, no __bf16 arithmetic
static __device__ __forceinline__ unsigned short f32_to_bf16_bits(float f) {
    unsigned int u = __builtin_bit_cast(unsigned int, f);
    u += 0x7fffu + ((u >> 16) & 1u);
    return (unsigned short)(u >> 16);
}

static __device__ __forceinline__ v8f wmma_bf16(v16bf a, v16bf b, v8f c) {
    return __builtin_amdgcn_wmma_f32_16x16x32_bf16(
               false, a, false, b, (short)0, c, false, false);
}

// ---------------------------------------------------------------------------
// K0: elementwise fp32 -> bf16 conversion (grid-stride)
// ---------------------------------------------------------------------------
__global__ void cvt_bf16_kernel(const float* __restrict__ s,
                                unsigned short* __restrict__ d, int n) {
    for (int i = blockIdx.x * blockDim.x + threadIdx.x; i < n;
         i += gridDim.x * blockDim.x)
        d[i] = f32_to_bf16_bits(s[i]);
}

// ---------------------------------------------------------------------------
// K1: bf16 GEMM, C[M,N] = A[M,K] * B[N,K]^T
//   One wave computes a 32x32 register tile: {a0,a1} x {b0,b1} -> 4 WMMAs
//   per K-slab of 32 (16 flops/byte vs 8 for the naive 16x16 scheme).
//   A row-major MxK bf16; Bw row-major NxK bf16 (weight rows == B columns).
// ---------------------------------------------------------------------------
__global__ void gemm_bf16_wmma_kernel(const unsigned short* __restrict__ A,
                                      const unsigned short* __restrict__ Bw,
                                      float* __restrict__ C,
                                      int M, int N, int K) {
    const int wid  = (blockIdx.x * blockDim.x + threadIdx.x) >> 5;
    const int lane = threadIdx.x & 31;
    const int tiles_n = N >> 5;               // 32-wide tiles
    const int tm = wid / tiles_n;
    const int tn = wid - tm * tiles_n;
    if (tm * 32 >= M) return;                 // wave-uniform guard

    const int r     = lane & 15;
    const int khalf = (lane >> 4) * 16;       // lanes 16..31 hold K=16..31 half

    const unsigned short* a0p = A  + (size_t)(tm * 32 + r) * K + khalf;
    const unsigned short* a1p = a0p + (size_t)16 * K;
    const unsigned short* b0p = Bw + (size_t)(tn * 32 + r) * K + khalf;
    const unsigned short* b1p = b0p + (size_t)16 * K;

    v8f c00 = {}, c01 = {}, c10 = {}, c11 = {};
    for (int k = 0; k < K; k += 32) {
        v16bf a0 = *reinterpret_cast<const v16bf*>(a0p + k);
        v16bf a1 = *reinterpret_cast<const v16bf*>(a1p + k);
        v16bf b0 = *reinterpret_cast<const v16bf*>(b0p + k);
        v16bf b1 = *reinterpret_cast<const v16bf*>(b1p + k);
        if (k + 32 < K) {                     // global_prefetch_b8 next K-slab
            __builtin_prefetch(a0p + k + 32, 0, 3);
            __builtin_prefetch(b0p + k + 32, 0, 3);
        }
        c00 = wmma_bf16(a0, b0, c00);
        c01 = wmma_bf16(a0, b1, c01);
        c10 = wmma_bf16(a1, b0, c10);
        c11 = wmma_bf16(a1, b1, c11);
    }

    // C/D layout: VGPR i -> M = i (lanes 0-15) or M = 8+i (lanes 16-31)
    const int orow = tm * 32 + ((lane >> 4) << 3);
    const int ocol = tn * 32 + r;
    float* cp = C + (size_t)orow * N + ocol;
#pragma unroll
    for (int i = 0; i < 8; ++i) {
        cp[(size_t)i * N]            = c00[i];
        cp[(size_t)i * N + 16]       = c01[i];
        cp[(size_t)(i + 16) * N]     = c10[i];
        cp[(size_t)(i + 16) * N + 16]= c11[i];
    }
}

// ---------------------------------------------------------------------------
// K2: depthwise conv(kernel 4, pad (1,2)) + bias + SiLU for x and z halves
//   xz: [MROWS, 1024] fp32. Emits xc fp32 (scan input), xc bf16 (x_proj A),
//   zc bf16 straight into columns [512..1024) of the out_proj concat buffer.
// ---------------------------------------------------------------------------
__global__ void conv_silu_kernel(const float* __restrict__ xz,
                                 const float* __restrict__ wx,
                                 const float* __restrict__ bx,
                                 const float* __restrict__ wz,
                                 const float* __restrict__ bz,
                                 float* __restrict__ xc_f32,
                                 unsigned short* __restrict__ xc_bf,
                                 unsigned short* __restrict__ concat_bf) {
    const size_t idx = (size_t)blockIdx.x * blockDim.x + threadIdx.x;
    const int d  = (int)(idx & (D_HALF - 1));
    const size_t bl = idx >> 9;               // b*SEQ + l
    const int l  = (int)(bl & (SEQ - 1));
    const size_t base = (bl - l) * (size_t)D_INNER;

    float accx = bx[d], accz = bz[d];
#pragma unroll
    for (int j = 0; j < D_CONV; ++j) {
        const int ll = l + j - 1;             // padding (1,2)
        if (ll >= 0 && ll < SEQ) {
            const float* p = xz + base + (size_t)ll * D_INNER;
            accx = fmaf(p[d],          wx[d * D_CONV + j], accx);
            accz = fmaf(p[D_HALF + d], wz[d * D_CONV + j], accz);
        }
    }
    const float sx = accx / (1.0f + __expf(-accx));   // silu
    const float sz = accz / (1.0f + __expf(-accz));
    xc_f32[idx] = sx;
    xc_bf[idx]  = f32_to_bf16_bits(sx);
    concat_bf[bl * (size_t)D_INNER + D_HALF + d] = f32_to_bf16_bits(sz);
}

// ---------------------------------------------------------------------------
// K3: LayerNorm over the 64-wide x_dbl rows (one wave per row, 2 cols/lane),
//     in-place fp32 update; also emits dt_raw (cols 0..31) as bf16.
// ---------------------------------------------------------------------------
__global__ void layernorm64_kernel(float* __restrict__ xdbl,
                                   const float* __restrict__ lw,
                                   const float* __restrict__ lb,
                                   unsigned short* __restrict__ dtraw_bf) {
    const int wave = threadIdx.x >> 5;
    const int lane = threadIdx.x & 31;
    const int rowi = blockIdx.x * (blockDim.x >> 5) + wave;
    float* rp = xdbl + (size_t)rowi * NDBL;

    const int c0 = lane * 2;
    float v0 = rp[c0], v1 = rp[c0 + 1];
    float s  = v0 + v1;
    float ss = v0 * v0 + v1 * v1;
#pragma unroll
    for (int off = 16; off >= 1; off >>= 1) {
        s  += __shfl_xor(s,  off, 32);
        ss += __shfl_xor(ss, off, 32);
    }
    const float mean = s * (1.0f / NDBL);
    const float var  = ss * (1.0f / NDBL) - mean * mean;
    const float rstd = rsqrtf(var + LN_EPS);

    const float n0 = (v0 - mean) * rstd * lw[c0]     + lb[c0];
    const float n1 = (v1 - mean) * rstd * lw[c0 + 1] + lb[c0 + 1];
    rp[c0]     = n0;
    rp[c0 + 1] = n1;
    if (c0 < DT_RANK) {                       // lanes 0..15 -> dt_raw bf16
        unsigned short* dp = dtraw_bf + (size_t)rowi * DT_RANK + c0;
        dp[0] = f32_to_bf16_bits(n0);
        dp[1] = f32_to_bf16_bits(n1);
    }
}

// ---------------------------------------------------------------------------
// K4: dt GEMM (K=32 -> single WMMA per tile) + clip + bias + softplus
//   delta[M,512] = softplus(clip(dt_raw[M,32] @ Wdt[512,32]^T, +-6) + dtb)
// ---------------------------------------------------------------------------
__global__ void dt_softplus_wmma_kernel(const unsigned short* __restrict__ dtraw,
                                        const unsigned short* __restrict__ Wdt,
                                        const float* __restrict__ dtb,
                                        float* __restrict__ delta) {
    const int N = D_HALF, K = DT_RANK;
    const int wid  = (blockIdx.x * blockDim.x + threadIdx.x) >> 5;
    const int lane = threadIdx.x & 31;
    const int tiles_n = N >> 4;
    const int tm = wid / tiles_n;
    const int tn = wid - tm * tiles_n;

    const int row   = tm * 16 + (lane & 15);
    const int col   = tn * 16 + (lane & 15);
    const int khalf = (lane >> 4) * 16;

    v16bf a = *reinterpret_cast<const v16bf*>(dtraw + (size_t)row * K + khalf);
    v16bf b = *reinterpret_cast<const v16bf*>(Wdt   + (size_t)col * K + khalf);
    v8f acc = {};
    acc = wmma_bf16(a, b, acc);

    const float bias = dtb[col];
    const int orow = tm * 16 + ((lane >> 4) << 3);
    float* dp = delta + (size_t)orow * N + col;
#pragma unroll
    for (int i = 0; i < 8; ++i) {
        float t = fminf(6.0f, fmaxf(-6.0f, acc[i])) + bias;
        dp[(size_t)i * N] = log1pf(__expf(t));        // softplus, t in [-6.3,6.3]
    }
}

// ---------------------------------------------------------------------------
// K5: selective scan. One lane owns one (b,d) pair; 16-state recurrence in
//     registers, 4096 sequential steps. 2048 lanes total = 64 waves.
//     B/C (shared by all 256 lanes of a block) staged into LDS in 64-step
//     chunks: one coalesced cooperative load + barrier per chunk, then
//     conflict-free ds_load broadcasts in the recurrence.
//     y -> bf16 into columns [0..512) of the out_proj concat buffer.
// ---------------------------------------------------------------------------
__global__ void selective_scan_kernel(const float* __restrict__ delta,
                                      const float* __restrict__ xc,
                                      const float* __restrict__ xdbl,   // post-LN
                                      const float* __restrict__ A_log,
                                      const float* __restrict__ Dp,
                                      unsigned short* __restrict__ concat_bf) {
    __shared__ float sBC[SCAN_CHUNK][2 * D_STATE];    // 8 KB

    const int b = blockIdx.x >> 1;
    const int d = ((blockIdx.x & 1) << 8) + threadIdx.x;

    float A[D_STATE], h[D_STATE];
#pragma unroll
    for (int n = 0; n < D_STATE; ++n) {
        A[n] = -__expf(A_log[d * D_STATE + n]);
        h[n] = 0.0f;
    }
    const float dscale = Dp[d];

    const size_t bl0 = (size_t)b * SEQ;
    for (int l0 = 0; l0 < SEQ; l0 += SCAN_CHUNK) {
        __syncthreads();                      // chunk buffer free to overwrite
        // cooperative coalesced load: SCAN_CHUNK*32 floats, 8 per thread
        for (int i = threadIdx.x; i < SCAN_CHUNK * 2 * D_STATE; i += 256) {
            const int step = i >> 5, c = i & 31;
            sBC[step][c] = xdbl[(bl0 + l0 + step) * NDBL + DT_RANK + c];
        }
        __syncthreads();

        size_t bl = bl0 + l0;
        for (int t = 0; t < SCAN_CHUNK; ++t, ++bl) {
            const float dt = delta[bl * D_HALF + d];
            const float u  = xc[bl * D_HALF + d];
            const float du = dt * u;
            float y = 0.0f;
#pragma unroll
            for (int n = 0; n < D_STATE; ++n) {
                h[n] = fmaf(__expf(dt * A[n]), h[n], du * sBC[t][n]);
                y    = fmaf(h[n], sBC[t][D_STATE + n], y);
            }
            y = fmaf(u, dscale, y);
            concat_bf[bl * (size_t)D_INNER + d] = f32_to_bf16_bits(y);
        }
    }
}

// ---------------------------------------------------------------------------
// Host launch
// ---------------------------------------------------------------------------
extern "C" void kernel_launch(void* const* d_in, const int* in_sizes, int n_in,
                              void* d_out, int out_size, void* d_ws, size_t ws_size,
                              hipStream_t stream) {
    const float* hs      = (const float*)d_in[0];   // (4,4096,512)
    const float* w_in    = (const float*)d_in[1];   // (1024,512)
    const float* w_xp    = (const float*)d_in[2];   // (64,512)
    const float* w_dt    = (const float*)d_in[3];   // (512,32)
    const float* b_dt    = (const float*)d_in[4];   // (512,)
    const float* A_log   = (const float*)d_in[5];   // (512,16)
    const float* D_param = (const float*)d_in[6];   // (512,)
    const float* cxw     = (const float*)d_in[7];   // (512,1,4)
    const float* cxb     = (const float*)d_in[8];   // (512,)
    const float* czw     = (const float*)d_in[9];   // (512,1,4)
    const float* czb     = (const float*)d_in[10];  // (512,)
    const float* ln_w    = (const float*)d_in[11];  // (64,)
    const float* ln_b    = (const float*)d_in[12];  // (64,)
    const float* w_out   = (const float*)d_in[13];  // (512,1024)
    float* out = (float*)d_out;                     // (4,4096,512) fp32

    // ---- workspace carve-up (all offsets 256B aligned)
    char* ws = (char*)d_ws;
    size_t off = 0;
    auto alloc = [&](size_t bytes) { void* p = ws + off; off = (off + bytes + 255) & ~(size_t)255; return p; };
    unsigned short* hs_bf   = (unsigned short*)alloc((size_t)MROWS * D_MODEL * 2);
    unsigned short* win_bf  = (unsigned short*)alloc((size_t)D_INNER * D_MODEL * 2);
    unsigned short* wxp_bf  = (unsigned short*)alloc((size_t)NDBL * D_HALF * 2);
    unsigned short* wdt_bf  = (unsigned short*)alloc((size_t)D_HALF * DT_RANK * 2);
    unsigned short* wout_bf = (unsigned short*)alloc((size_t)D_MODEL * D_INNER * 2);
    float*          xz      = (float*)alloc((size_t)MROWS * D_INNER * 4);
    float*          xc_f32  = (float*)alloc((size_t)MROWS * D_HALF * 4);
    unsigned short* xc_bf   = (unsigned short*)alloc((size_t)MROWS * D_HALF * 2);
    unsigned short* cat_bf  = (unsigned short*)alloc((size_t)MROWS * D_INNER * 2);
    float*          xdbl    = (float*)alloc((size_t)MROWS * NDBL * 4);
    unsigned short* dtraw   = (unsigned short*)alloc((size_t)MROWS * DT_RANK * 2);
    float*          delta   = (float*)alloc((size_t)MROWS * D_HALF * 4);
    (void)ws_size; (void)n_in; (void)in_sizes; (void)out_size;

    const int T = 256;
    auto cvt = [&](const float* s, unsigned short* d, int n) {
        cvt_bf16_kernel<<<(n + T - 1) / T, T, 0, stream>>>(s, d, n);
    };
    // ---- fp32 -> bf16 staging
    cvt(hs,    hs_bf,   MROWS * D_MODEL);
    cvt(w_in,  win_bf,  D_INNER * D_MODEL);
    cvt(w_xp,  wxp_bf,  NDBL * D_HALF);
    cvt(w_dt,  wdt_bf,  D_HALF * DT_RANK);
    cvt(w_out, wout_bf, D_MODEL * D_INNER);

    // ---- in_proj: xz = hs @ w_in^T   (16384 x 1024 x 512)
    {
        int tiles = (MROWS / 32) * (D_INNER / 32);   // 32x32 tiles per wave
        gemm_bf16_wmma_kernel<<<tiles / 8, T, 0, stream>>>(
            hs_bf, win_bf, xz, MROWS, D_INNER, D_MODEL);
    }
    // ---- depthwise conv + SiLU for x and z
    conv_silu_kernel<<<(MROWS * D_HALF) / T, T, 0, stream>>>(
        xz, cxw, cxb, czw, czb, xc_f32, xc_bf, cat_bf);

    // ---- x_proj: x_dbl = xc @ w_xp^T  (16384 x 64 x 512)
    {
        int tiles = (MROWS / 32) * (NDBL / 32);
        gemm_bf16_wmma_kernel<<<tiles / 8, T, 0, stream>>>(
            xc_bf, wxp_bf, xdbl, MROWS, NDBL, D_HALF);
    }
    // ---- LayerNorm(64) in place + dt_raw bf16
    layernorm64_kernel<<<MROWS / 8, T, 0, stream>>>(xdbl, ln_w, ln_b, dtraw);

    // ---- dt_proj (K=32, one WMMA/tile) + clip + softplus -> delta
    {
        int tiles = (MROWS / 16) * (D_HALF / 16);
        dt_softplus_wmma_kernel<<<tiles / 8, T, 0, stream>>>(
            dtraw, wdt_bf, b_dt, delta);
    }
    // ---- selective scan (sequential over L; 2048 parallel (b,d) lanes)
    selective_scan_kernel<<<B_SZ * 2, T, 0, stream>>>(
        delta, xc_f32, xdbl, A_log, D_param, cat_bf);

    // ---- out_proj: out = [y | zc] @ w_out^T  (16384 x 512 x 1024)
    {
        int tiles = (MROWS / 32) * (D_MODEL / 32);
        gemm_bf16_wmma_kernel<<<tiles / 8, T, 0, stream>>>(
            cat_bf, wout_bf, out, MROWS, D_MODEL, D_INNER);
    }
}
